// Model_24240795419597
// MI455X (gfx1250) — matrix operand order repaired and checked by reading, not compiled, not run on hardware
//
#include <hip/hip_runtime.h>
#include <hip/hip_bf16.h>

// ---------------------------------------------------------------------------
// Problem dims (fixed by the reference): B=16, T=512, V=8192, E=512, H=1024
// ---------------------------------------------------------------------------
#define DIM_B 16
#define DIM_T 512
#define DIM_V 8192
#define DIM_E 512
#define DIM_H 1024
#define DIM_4H 4096
#define ROWS (DIM_B * DIM_T)   // 8192

typedef __attribute__((ext_vector_type(16))) __bf16 bf16x16;
typedef __attribute__((ext_vector_type(8)))  float  f32x8;

// ---------------------------------------------------------------------------
// bf16 helpers (storage type = unsigned short)
// ---------------------------------------------------------------------------
__device__ __forceinline__ unsigned short f2bf(float f) {
    union { float f; unsigned int u; } v; v.f = f;
    if ((v.u & 0x7fffffffu) > 0x7f800000u) return (unsigned short)((v.u >> 16) | 0x40);
    unsigned int r = v.u + 0x7fffu + ((v.u >> 16) & 1u);
    return (unsigned short)(r >> 16);
}

// ---------------------------------------------------------------------------
// CDNA5 async global->LDS copy (ASYNCcnt path, bypasses VGPRs).
// vdst operand is the wave-relative LDS byte address; for a generic pointer
// into __shared__, that is addr[31:0] (ISA 10.2 aperture mapping).
// ---------------------------------------------------------------------------
__device__ __forceinline__ void async_copy_b128(unsigned int lds_off, const void* gptr) {
    asm volatile("global_load_async_to_lds_b128 %0, %1, off"
                 :: "v"(lds_off), "v"(gptr) : "memory");
}
__device__ __forceinline__ void wait_asynccnt0() {
    asm volatile("s_wait_asynccnt 0x0" ::: "memory");
}
__device__ __forceinline__ unsigned int lds_offset_of(const void* p) {
    return (unsigned int)(size_t)p;   // low 32 bits of flat addr == LDS offset
}

// ---------------------------------------------------------------------------
// WMMA fragment loaders (CDNA5 ISA VGPR layouts).
// A 16x32 bf16 fragment: lane half h = lane>>4,
//   elements 0..7  <- K = kk + 8*h + (0..7)      (one 16B load)
//   elements 8..15 <- K = kk + 16 + 8*h + (0..7) (one 16B load)
// B 32x16 bf16 fragment (loaded from W^T rows = columns of B):
//   elements 0..7 <- K = kk + 8*h + (0..7), 8..15 <- K = kk + 16 + 8*h + (0..7)
// ---------------------------------------------------------------------------
__device__ __forceinline__ bf16x16 frag_a(const unsigned short* rowptr, int kk, int lane) {
    union { bf16x16 v; uint4 q[2]; } f;
    const unsigned short* p = rowptr + kk + ((lane >> 4) << 3);
    f.q[0] = *(const uint4*)(p);
    f.q[1] = *(const uint4*)(p + 16);
    return f.v;
}
__device__ __forceinline__ bf16x16 frag_b(const unsigned short* rowptr, int kk, int lane) {
    union { bf16x16 v; uint4 q[2]; } f;
    const unsigned short* p = rowptr + kk + ((lane >> 4) << 3);
    f.q[0] = *(const uint4*)(p);
    f.q[1] = *(const uint4*)(p + 16);
    return f.v;
}
__device__ __forceinline__ f32x8 wmma_bf16(bf16x16 a, bf16x16 b, f32x8 c) {
    return __builtin_amdgcn_wmma_f32_16x16x32_bf16(
        false, a, false, b, (short)0, c, false, false);
}

__device__ __forceinline__ float sigmoidf(float x) { return 1.0f / (1.0f + __expf(-x)); }

// ---------------------------------------------------------------------------
// Kernel 1: column-wise weight norm + scale, output transposed bf16.
//   w : [R, C] f32 row-major, g : [C] -> wT : [C, R] bf16
// ---------------------------------------------------------------------------
__global__ __launch_bounds__(256) void colnorm_kernel(
    const float* __restrict__ w, const float* __restrict__ g,
    unsigned short* __restrict__ wT, int R, int C) {
    __shared__ float red[256];
    const int c = blockIdx.x, tid = threadIdx.x;
    float ss = 0.f;
    for (int r = tid; r < R; r += 256) {
        float x = w[(size_t)r * C + c];
        ss += x * x;
    }
    red[tid] = ss;
    __syncthreads();
    for (int s = 128; s > 0; s >>= 1) {
        if (tid < s) red[tid] += red[tid + s];
        __syncthreads();
    }
    const float scale = g[c] / fmaxf(sqrtf(red[0]), 1e-12f);
    for (int r = tid; r < R; r += 256)
        wT[(size_t)c * R + r] = f2bf(w[(size_t)r * C + c] * scale);
}

// ---------------------------------------------------------------------------
// Kernel 2: f32 -> bf16 conversion (pred_w is already [V, H] = [N, K])
// ---------------------------------------------------------------------------
__global__ __launch_bounds__(256) void cvt_bf16_kernel(
    const float* __restrict__ src, unsigned short* __restrict__ dst, int n) {
    int i = blockIdx.x * 256 + threadIdx.x;
    if (i < n) dst[i] = f2bf(src[i]);
}

// ---------------------------------------------------------------------------
// Kernel 3: embedding gather (padding_idx = 0) -> bf16 activations [ROWS, E]
// ---------------------------------------------------------------------------
__global__ __launch_bounds__(256) void embed_kernel(
    const int* __restrict__ xs, const float* __restrict__ ew,
    unsigned short* __restrict__ xo) {
    int i = blockIdx.x * 256 + threadIdx.x;
    int row = i >> 9;
    int e   = i & (DIM_E - 1);
    int idx = xs[row];
    float v = (idx == 0) ? 0.f : ew[(size_t)idx * DIM_E + e];
    xo[i] = f2bf(v);
}

// ---------------------------------------------------------------------------
// Kernel 4: input-projection GEMM, K = E = 512 fixed.
//   C[M,N] = A[M,512](bf16) x B^T[N,512](bf16) + bias
// Block = 256 threads = 8 waves; BM = 64 (4 row-tiles/wave share one B frag
// -> 4x arithmetic intensity on the B stream), BN = 128.
// A tile (64 x 512 bf16 = 64 KB) staged in LDS via async-to-LDS DMA.
// grid = (N/128, M/64).
// ---------------------------------------------------------------------------
__global__ __launch_bounds__(256) void gemm_bf16_kernel(
    const unsigned short* __restrict__ A, const unsigned short* __restrict__ Bt,
    const float* __restrict__ bias, float* __restrict__ C, int N) {
    __shared__ __align__(16) unsigned short As[64 * DIM_E];   // 64 KB
    const int tid = threadIdx.x, lane = tid & 31, wave = tid >> 5;
    const int m0 = blockIdx.y * 64;
    const int n0 = blockIdx.x * 128 + wave * 16;

    // Async-stage the contiguous 64 x 512 A tile into LDS (16B per op).
    {
        const char* gsrc = (const char*)(A + (size_t)m0 * DIM_E);
        const unsigned int lbase = lds_offset_of(As);
        for (int i = tid; i < (64 * DIM_E * 2) / 16; i += 256)
            async_copy_b128(lbase + i * 16, gsrc + (size_t)i * 16);
        wait_asynccnt0();
    }
    __syncthreads();

    f32x8 acc[4] = {{}, {}, {}, {}};
    const unsigned short* brow = Bt + (size_t)(n0 + (lane & 15)) * DIM_E;
    for (int kk = 0; kk < DIM_E; kk += 32) {
        bf16x16 b = frag_b(brow, kk, lane);
#pragma unroll
        for (int mt = 0; mt < 4; ++mt) {
            const unsigned short* arow = &As[(mt * 16 + (lane & 15)) * DIM_E];
            bf16x16 a = frag_a(arow, kk, lane);
            acc[mt] = wmma_bf16(a, b, acc[mt]);
        }
    }

    const int col = n0 + (lane & 15);
    const float bv = bias ? bias[col] : 0.f;
    const int rowbase = ((lane >> 4) << 3);
#pragma unroll
    for (int mt = 0; mt < 4; ++mt) {
        const int rb = m0 + mt * 16 + rowbase;
#pragma unroll
        for (int v = 0; v < 8; ++v)
            C[(size_t)(rb + v) * N + col] = acc[mt][v] + bv;
    }
}

// ---------------------------------------------------------------------------
// Kernel 5: zero the scalar output
// ---------------------------------------------------------------------------
__global__ void zero_kernel(float* out) { out[0] = 0.f; }

// ---------------------------------------------------------------------------
// Kernel 6: mLSTM recurrence, single workgroup of 1024 threads (32 waves).
// h (bf16) and m (bf16) live in LDS; f32 cell state c stays pinned in
// registers per wave (wave w owns h-column tiles 2w, 2w+1 for all 512 steps).
// ---------------------------------------------------------------------------
__global__ __launch_bounds__(1024) void mlstm_recurrence_kernel(
    const float* __restrict__ zx, const float* __restrict__ mx,
    const unsigned short* __restrict__ whT, const unsigned short* __restrict__ wmhT,
    unsigned short* __restrict__ hs) {
    __shared__ __align__(16) unsigned short hS[DIM_B * DIM_H];  // 32 KB
    __shared__ __align__(16) unsigned short mS[DIM_B * DIM_H];  // 32 KB
    const int tid = threadIdx.x, lane = tid & 31, wave = tid >> 5;
    const int colbase = (lane & 15);
    const int rowbase = ((lane >> 4) << 3);

    for (int i = tid; i < (DIM_B * DIM_H) / 2; i += 1024) ((unsigned int*)hS)[i] = 0u;
    __syncthreads();

    f32x8 cst[2] = {{}, {}};

    for (int t = 0; t < DIM_T; ++t) {
        // phase 1: hm = h @ wmh_n ; m = mx_t * hm -> LDS
#pragma unroll
        for (int jj = 0; jj < 2; ++jj) {
            const int j  = 2 * wave + jj;
            const int n0 = j * 16;
            f32x8 acc = {};
            const unsigned short* arow = &hS[colbase * DIM_H];
            const unsigned short* brow = wmhT + (size_t)(n0 + colbase) * DIM_H;
            for (int kk = 0; kk < DIM_H; kk += 32) {
                bf16x16 a = frag_a(arow, kk, lane);
                bf16x16 b = frag_b(brow, kk, lane);
                acc = wmma_bf16(a, b, acc);
            }
            const int col = n0 + colbase;
#pragma unroll
            for (int v = 0; v < 8; ++v) {
                const int bb = rowbase + v;
                float mval = mx[((size_t)bb * DIM_T + t) * DIM_H + col] * acc[v];
                mS[bb * DIM_H + col] = f2bf(mval);
            }
        }
        __syncthreads();

        // phase 2: z = zx_t + m @ wh_n ; gates
#pragma unroll
        for (int jj = 0; jj < 2; ++jj) {
            const int j = 2 * wave + jj;
            f32x8 g[4];
#pragma unroll
            for (int gi = 0; gi < 4; ++gi) {
                const int n0 = gi * DIM_H + j * 16;
                f32x8 acc = {};
                const unsigned short* arow = &mS[colbase * DIM_H];
                const unsigned short* brow = whT + (size_t)(n0 + colbase) * DIM_H;
                for (int kk = 0; kk < DIM_H; kk += 32) {
                    bf16x16 a = frag_a(arow, kk, lane);
                    bf16x16 b = frag_b(brow, kk, lane);
                    acc = wmma_bf16(a, b, acc);
                }
                g[gi] = acc;
            }
            const int col = j * 16 + colbase;
#pragma unroll
            for (int v = 0; v < 8; ++v) {
                const int bb = rowbase + v;
                const size_t zr = ((size_t)bb * DIM_T + t) * DIM_4H;
                float iv = sigmoidf(g[0][v] + zx[zr + 0 * DIM_H + col]);
                float fv = sigmoidf(g[1][v] + zx[zr + 1 * DIM_H + col]);
                float ov = sigmoidf(g[2][v] + zx[zr + 2 * DIM_H + col]);
                float uv = tanhf  (g[3][v] + zx[zr + 3 * DIM_H + col]);
                float cn = fv * cst[jj][v] + iv * uv;
                cst[jj][v] = cn;
                float hv = ov * tanhf(cn);
                unsigned short hb = f2bf(hv);
                hS[bb * DIM_H + col] = hb;
                hs[((size_t)bb * DIM_T + t) * DIM_H + col] = hb;
            }
        }
        __syncthreads();
    }
}

// ---------------------------------------------------------------------------
// Kernel 7: fused logits GEMM + online log-softmax + NLL mean.
// One block per 64 rows (128 blocks); 8 waves sweep V in 128-col chunks,
// each wave computing 4 row-tiles that share one pred_w fragment
// (4x reuse => pred_w L2 traffic 8 GB -> 2 GB, balancing the roofline).
// hs A-tile (64 x 1024 bf16 = 128 KB) staged via async-to-LDS DMA.
// ---------------------------------------------------------------------------
__global__ __launch_bounds__(256) void logits_nll_kernel(
    const unsigned short* __restrict__ hs, const unsigned short* __restrict__ pw,
    const float* __restrict__ pb, const int* __restrict__ ys,
    float* __restrict__ out) {
    __shared__ __align__(16) unsigned short hA[64 * DIM_H];   // 128 KB
    __shared__ float buf[64 * 128];                           // 32 KB
    __shared__ float rmax[64], rsum[64], rtgt[64];
    const int tid = threadIdx.x, lane = tid & 31, wave = tid >> 5;
    const int r0 = blockIdx.x * 64;

    // Async-stage the 64 x H bf16 A tile into LDS.
    {
        const char* gsrc = (const char*)(hs + (size_t)r0 * DIM_H);
        const unsigned int lbase = lds_offset_of(hA);
        for (int i = tid; i < (64 * DIM_H * 2) / 16; i += 256)
            async_copy_b128(lbase + i * 16, gsrc + (size_t)i * 16);
        wait_asynccnt0();
    }
    if (tid < 64) { rmax[tid] = -1e30f; rsum[tid] = 0.f; rtgt[tid] = 0.f; }
    __syncthreads();

    for (int ch = 0; ch < DIM_V / 128; ++ch) {
        const int n0 = ch * 128 + wave * 16;
        const unsigned short* brow = pw + (size_t)(n0 + (lane & 15)) * DIM_H;
        // prefetch next chunk's pred_w row into L2 (global_prefetch_b8)
        if (ch + 1 < DIM_V / 128)
            __builtin_prefetch(brow + (size_t)128 * DIM_H, 0, 1);

        f32x8 acc[4] = {{}, {}, {}, {}};
        for (int kk = 0; kk < DIM_H; kk += 32) {
            bf16x16 b = frag_b(brow, kk, lane);
#pragma unroll
            for (int mt = 0; mt < 4; ++mt) {
                const unsigned short* arow = &hA[(mt * 16 + (lane & 15)) * DIM_H];
                bf16x16 a = frag_a(arow, kk, lane);
                acc[mt] = wmma_bf16(a, b, acc[mt]);
            }
        }
        const float bv = pb[n0 + (lane & 15)];
        const int lcol = wave * 16 + (lane & 15);
        const int rowbase = ((lane >> 4) << 3);
#pragma unroll
        for (int mt = 0; mt < 4; ++mt)
#pragma unroll
            for (int v = 0; v < 8; ++v)
                buf[(mt * 16 + rowbase + v) * 128 + lcol] = acc[mt][v] + bv;
        __syncthreads();

        if (tid < 64) {
            const int r = tid;
            const int tgt = ys[r0 + r];
            float m = rmax[r], s = rsum[r], tg = rtgt[r];
            float cm = -1e30f;
#pragma unroll 1
            for (int cj = 0; cj < 128; ++cj) cm = fmaxf(cm, buf[r * 128 + cj]);
            const float nm = fmaxf(m, cm);
            s *= __expf(m - nm);
#pragma unroll 1
            for (int cj = 0; cj < 128; ++cj) {
                const float x = buf[r * 128 + cj];
                s += __expf(x - nm);
                if (ch * 128 + cj == tgt) tg = x;
            }
            rmax[r] = nm; rsum[r] = s; rtgt[r] = tg;
        }
        __syncthreads();
    }

    if (tid < 64) {
        const float nll = (rmax[tid] + __logf(rsum[tid])) - rtgt[tid];
        atomicAdd(out, nll * (1.0f / (float)ROWS));
    }
}

// ---------------------------------------------------------------------------
// Host launcher
// ---------------------------------------------------------------------------
extern "C" void kernel_launch(void* const* d_in, const int* in_sizes, int n_in,
                              void* d_out, int out_size, void* d_ws, size_t ws_size,
                              hipStream_t stream) {
    (void)in_sizes; (void)n_in; (void)out_size; (void)ws_size;
    const int*   xs      = (const int*)  d_in[0];
    const int*   ys      = (const int*)  d_in[1];
    const float* embed_w = (const float*)d_in[2];
    const float* wx      = (const float*)d_in[3];
    const float* wh      = (const float*)d_in[4];
    const float* wmx     = (const float*)d_in[5];
    const float* wmh     = (const float*)d_in[6];
    const float* b       = (const float*)d_in[7];
    const float* gx      = (const float*)d_in[8];
    const float* gh      = (const float*)d_in[9];
    const float* gmx     = (const float*)d_in[10];
    const float* gmh     = (const float*)d_in[11];
    const float* pred_w  = (const float*)d_in[12];
    const float* pred_b  = (const float*)d_in[13];
    float* out = (float*)d_out;

    char* w = (char*)d_ws;
    unsigned short* x_bf  = (unsigned short*)w; w += (size_t)ROWS  * DIM_E  * 2; // 8 MB
    unsigned short* wxT   = (unsigned short*)w; w += (size_t)DIM_4H* DIM_E  * 2; // 4 MB
    unsigned short* whT   = (unsigned short*)w; w += (size_t)DIM_4H* DIM_H  * 2; // 8 MB
    unsigned short* wmxT  = (unsigned short*)w; w += (size_t)DIM_H * DIM_E  * 2; // 1 MB
    unsigned short* wmhT  = (unsigned short*)w; w += (size_t)DIM_H * DIM_H  * 2; // 2 MB
    unsigned short* pw_bf = (unsigned short*)w; w += (size_t)DIM_V * DIM_H  * 2; // 16 MB
    float*          zx    = (float*)w;          w += (size_t)ROWS  * DIM_4H * 4; // 128 MB
    float*          mx    = (float*)w;          w += (size_t)ROWS  * DIM_H  * 4; // 32 MB
    unsigned short* hs    = (unsigned short*)w; w += (size_t)ROWS  * DIM_H  * 2; // 16 MB

    colnorm_kernel<<<DIM_4H, 256, 0, stream>>>(wx,  gx,  wxT,  DIM_E, DIM_4H);
    colnorm_kernel<<<DIM_4H, 256, 0, stream>>>(wh,  gh,  whT,  DIM_H, DIM_4H);
    colnorm_kernel<<<DIM_H,  256, 0, stream>>>(wmx, gmx, wmxT, DIM_E, DIM_H);
    colnorm_kernel<<<DIM_H,  256, 0, stream>>>(wmh, gmh, wmhT, DIM_H, DIM_H);

    {
        const int n = DIM_V * DIM_H;
        cvt_bf16_kernel<<<n / 256, 256, 0, stream>>>(pred_w, pw_bf, n);
    }

    embed_kernel<<<(ROWS * DIM_E) / 256, 256, 0, stream>>>(xs, embed_w, x_bf);

    // zx = x @ wx_n + b   [8192, 4096], K = 512
    gemm_bf16_kernel<<<dim3(DIM_4H / 128, ROWS / 64), 256, 0, stream>>>(
        x_bf, wxT, b, zx, DIM_4H);
    // mx = x @ wmx_n      [8192, 1024], K = 512
    gemm_bf16_kernel<<<dim3(DIM_H / 128, ROWS / 64), 256, 0, stream>>>(
        x_bf, wmxT, nullptr, mx, DIM_H);

    zero_kernel<<<1, 1, 0, stream>>>(out);

    mlstm_recurrence_kernel<<<1, 1024, 0, stream>>>(zx, mx, whT, wmhT, hs);

    logits_nll_kernel<<<ROWS / 64, 256, 0, stream>>>(hs, pw_bf, pred_b, ys, out);
}
